// AMPBlock2_4140348473781
// MI455X (gfx1250) — matrix operand order repaired and checked by reading, not compiled, not run on hardware
//
#include <hip/hip_runtime.h>
#include <stdint.h>

// ---------------------------------------------------------------------------
// AMP block x3 for MI455X (gfx1250, wave32, WMMA).
//   B=8, C=512, T=8192, KF=12, RATIO=2, dilations {1,3,5}
// Pipeline per block:
//   act512  : fused edge-pad -> 2x upsample (12-tap) -> snake -> 12-tap
//             downsample; writes bf16 xt in [b][t][c] layout (LDS transpose)
//   wprep   : weight-norm -> bf16 W in [tap][o][i] layout (WMMA A layout)
//   gemm512 : 512x512x(3*512) GEMM via v_wmma_f32_16x16x32_bf16 + bias + residual
// ---------------------------------------------------------------------------

#define TT   8192
#define CC   512
#define BB   8
#define TPAD 8208   // 8 zero rows each side so dilated taps (|off|<=5) never predicate

typedef __attribute__((ext_vector_type(16))) __bf16 v16bf;
typedef __attribute__((ext_vector_type(8)))  float  v8f;

union V16 { uint4 q[2]; v16bf v; };

__device__ __forceinline__ uint16_t f2bf(float f) {
  uint32_t u = __float_as_uint(f);
  return (uint16_t)((u + 0x7fffu + ((u >> 16) & 1u)) >> 16);   // RNE
}

// ---------------- Kaiser-sinc filter, computed once on device --------------
__device__ double i0d(double x) {
  double t2 = 0.25 * x * x, term = 1.0, sum = 1.0;
  for (int k = 1; k < 40; ++k) { term *= t2 / (double)(k * k); sum += term; }
  return sum;
}

__global__ void filt_init(float* __restrict__ filt) {
  if (threadIdx.x != 0 || blockIdx.x != 0) return;
  const double PI = 3.14159265358979323846;
  const double cutoff = 0.25, half_width = 0.3;
  const double delta_f = 4.0 * half_width;
  const double A = 2.285 * 5.0 * PI * delta_f + 7.95;
  double beta;
  if (A > 50.0)      beta = 0.1102 * (A - 8.7);
  else if (A >= 21.0) beta = 0.5842 * pow(A - 21.0, 0.4) + 0.07886 * (A - 21.0);
  else                beta = 0.0;
  const double i0b = i0d(beta);
  double w[12], s = 0.0;
  for (int n = 0; n < 12; ++n) {
    double ratio = ((double)n - 5.5) / 5.5;                 // np.kaiser window arg
    double r2 = 1.0 - ratio * ratio; if (r2 < 0.0) r2 = 0.0;
    double win = i0d(beta * sqrt(r2)) / i0b;
    double time = ((double)n - 6.0) + 0.5;                  // -5.5 .. 5.5
    double sx = 2.0 * cutoff * time;
    double sinc = (sx == 0.0) ? 1.0 : sin(PI * sx) / (PI * sx);
    w[n] = 2.0 * cutoff * win * sinc;
    s += w[n];
  }
  for (int n = 0; n < 12; ++n) filt[n] = (float)(w[n] / s);
}

// ---------------- zero the time padding rows of xt -------------------------
__global__ void padzero(uint16_t* __restrict__ xt) {
  int i = blockIdx.x * blockDim.x + threadIdx.x;
  if (i >= BB * 16 * CC) return;
  int b = i >> 13, rem = i & 8191;
  int r = rem >> 9, c = rem & 511;
  int row = (r < 8) ? r : (8192 + r);        // rows 0..7 and 8200..8207
  xt[((size_t)b * TPAD + row) * CC + c] = 0;
}

// ---------------- weight-norm + bf16 repack: Wb[tap][o][i] -----------------
__global__ __launch_bounds__(256) void wprep(const float* __restrict__ v,
                                             const float* __restrict__ g,
                                             uint16_t* __restrict__ Wb) {
  __shared__ float red[256];
  const int o = blockIdx.x, tid = threadIdx.x;
  const float* vo = v + (size_t)o * (CC * 3);
  float s = 0.f;
  for (int j = tid; j < CC * 3; j += 256) { float t = vo[j]; s += t * t; }
  red[tid] = s; __syncthreads();
  for (int w = 128; w > 0; w >>= 1) { if (tid < w) red[tid] += red[tid + w]; __syncthreads(); }
  const float scale = g[o] * rsqrtf(red[0]);
  for (int j = tid; j < CC * 3; j += 256) {
    int c = j / 3, k = j - 3 * c;                       // v layout [o][c][k]
    Wb[((size_t)k * CC + o) * CC + c] = f2bf(vo[j] * scale);
  }
}

// ---------------- fused upsample->snake->downsample ------------------------
// Each block: 32 channels x 32 output times for one batch. Output layout
// xt[b][t+8][c] (bf16, time-major rows) via an LDS transpose so stores coalesce.
__global__ __launch_bounds__(256) void act512(const float* __restrict__ x,
                                              const float* __restrict__ alpha,
                                              const float* __restrict__ beta,
                                              const float* __restrict__ filt,
                                              uint16_t* __restrict__ xt) {
  __shared__ float xs[32][44];   // x[s0-5 .. s0+36] per channel row (42 used)
  __shared__ float zt[32][33];   // [c][s] staging for transpose
  const int tid = threadIdx.x;
  const int s0 = blockIdx.x * 32, c0 = blockIdx.y * 32, b = blockIdx.z;

  float f[12];
#pragma unroll
  for (int k = 0; k < 12; ++k) f[k] = filt[k];

  for (int idx = tid; idx < 32 * 42; idx += 256) {
    int r = idx / 42, j = idx - 42 * r;
    int sidx = s0 - 5 + j;
    sidx = sidx < 0 ? 0 : (sidx > TT - 1 ? TT - 1 : sidx);   // edge pad
    xs[r][j] = x[((size_t)b * CC + c0 + r) * TT + sidx];
  }
  __syncthreads();

  const int cl = tid >> 3;             // channel within tile
  const int sb = (tid & 7) * 4;        // 4 time outputs per thread
  const float a    = __expf(alpha[c0 + cl]);
  const float invb = 1.0f / (__expf(beta[c0 + cl]) + 1e-9f);
  const float* row = xs[cl];

#pragma unroll
  for (int q = 0; q < 4; ++q) {
    const int s = s0 + sb + q;
    float z = 0.f;
#pragma unroll
    for (int k = 0; k < 12; ++k) {
      int t = 2 * s + k - 5;                                  // upsampled index
      t = t < 0 ? 0 : (t > 2 * TT - 1 ? 2 * TT - 1 : t);      // edge pad at 2x rate
      const int fo = t & 1;
      const int j0 = (t >> 1) + 2 + fo - s0;                  // index into xs row
      float u = 0.f;
#pragma unroll
      for (int m = 0; m < 6; ++m) u += f[2 * m + fo] * row[j0 + m];
      u *= 2.f;                                               // RATIO
      const float sn = __sinf(u * a);
      const float vv = u + invb * sn * sn;                    // snake
      z += f[k] * vv;                                         // downsample tap
    }
    zt[cl][sb + q] = z;
  }
  __syncthreads();

  const int sw = tid >> 3;             // time row
  const int cw = (tid & 7) * 4;        // 4 channels -> one 8B store
  uint32_t lo = (uint32_t)f2bf(zt[cw + 0][sw]) | ((uint32_t)f2bf(zt[cw + 1][sw]) << 16);
  uint32_t hi = (uint32_t)f2bf(zt[cw + 2][sw]) | ((uint32_t)f2bf(zt[cw + 3][sw]) << 16);
  *(uint2*)(xt + ((size_t)b * TPAD + (s0 + sw + 8)) * CC + c0 + cw) = make_uint2(lo, hi);
}

// ---------------- WMMA GEMM: out[b,o,t] = sum_k W_k @ xt[.,t+(k-1)d] + bias + res
// Workgroup: 8 waves, each M=64 x N=32 (full M=512 per workgroup, N tile 32).
__global__ __launch_bounds__(256) void gemm512(const __bf16* __restrict__ Wb,
                                               const __bf16* __restrict__ xt,
                                               const float* __restrict__ xin,
                                               const float* __restrict__ bias,
                                               float* __restrict__ out, int dil) {
  const int lane = threadIdx.x & 31;
  const int wave = threadIdx.x >> 5;
  const int b  = blockIdx.z;
  const int tb = blockIdx.x * 32;
  const int obase = wave * 64;
  const int lh = lane >> 4;      // lane half selects contiguous K half: k = lh*16 + s
  const int ln = lane & 15;

  v8f acc[4][2] = {};

  for (int k = 0; k < 3; ++k) {
    const int off = (k - 1) * dil;
    // B fragment: xt is [b][t][c]; lane ln -> column t, lane half -> c+16
    const __bf16* brow0 = xt + ((size_t)b * TPAD + (tb + ln + off + 8)) * CC + lh * 16;
    const __bf16* brow1 = brow0 + 16 * CC;
    const __bf16* arow[4];
#pragma unroll
    for (int mi = 0; mi < 4; ++mi)
      arow[mi] = Wb + ((size_t)k * CC + obase + mi * 16 + ln) * CC + lh * 16;

#pragma unroll 1
    for (int ic = 0; ic < CC; ic += 32) {
      V16 B0, B1, A[4];
      B0.q[0] = *(const uint4*)(brow0 + ic);      // K = ic+lh*16 .. +7
      B0.q[1] = *(const uint4*)(brow0 + ic + 8);  // K = .. +15
      B1.q[0] = *(const uint4*)(brow1 + ic);
      B1.q[1] = *(const uint4*)(brow1 + ic + 8);
#pragma unroll
      for (int mi = 0; mi < 4; ++mi) {
        A[mi].q[0] = *(const uint4*)(arow[mi] + ic);        // K 0-7 / 16-23
        A[mi].q[1] = *(const uint4*)(arow[mi] + ic + 8);    // K 8-15 / 24-31
      }
#pragma unroll
      for (int mi = 0; mi < 4; ++mi) {
        acc[mi][0] = __builtin_amdgcn_wmma_f32_16x16x32_bf16(
            false, A[mi].v, false, B0.v, (short)0, acc[mi][0], false, false);
        acc[mi][1] = __builtin_amdgcn_wmma_f32_16x16x32_bf16(
            false, A[mi].v, false, B1.v, (short)0, acc[mi][1], false, false);
      }
    }
  }

  // C/D layout: lane column n=ln, VGPR r holds M = r + lh*8
#pragma unroll
  for (int mi = 0; mi < 4; ++mi)
#pragma unroll
    for (int ni = 0; ni < 2; ++ni) {
      const int t = tb + ni * 16 + ln;
#pragma unroll
      for (int r = 0; r < 8; ++r) {
        const int o = obase + mi * 16 + lh * 8 + r;
        const size_t idx = ((size_t)b * CC + o) * TT + t;
        out[idx] = acc[mi][ni][r] + bias[o] + xin[idx];
      }
    }
}

// ---------------------------------------------------------------------------
extern "C" void kernel_launch(void* const* d_in, const int* in_sizes, int n_in,
                              void* d_out, int out_size, void* d_ws, size_t ws_size,
                              hipStream_t stream) {
  const float* x  = (const float*)d_in[0];
  const float* cv = (const float*)d_in[1];   // (3,512,512,3)
  const float* cg = (const float*)d_in[2];   // (3,512,1,1)
  const float* cb = (const float*)d_in[3];   // (3,512)
  const float* al = (const float*)d_in[4];   // (3,512)
  const float* be = (const float*)d_in[5];   // (3,512)
  float* out = (float*)d_out;

  float*    filt = (float*)d_ws;
  uint16_t* Wb   = (uint16_t*)((char*)d_ws + 256);
  uint16_t* xt   = (uint16_t*)((char*)d_ws + 256 + (size_t)3 * CC * CC * 2);
  // ws usage: 256 + 1.5MB + 8*8208*512*2 B  ~= 66 MiB

  filt_init<<<1, 64, 0, stream>>>(filt);
  padzero<<<256, 256, 0, stream>>>(xt);

  const int dils[3] = {1, 3, 5};
  for (int i = 0; i < 3; ++i) {
    const float* xi = (i == 0) ? x : out;    // blocks 1,2 run in-place on d_out
    wprep<<<512, 256, 0, stream>>>(cv + (size_t)i * CC * CC * 3, cg + i * CC, Wb);
    act512<<<dim3(TT / 32, CC / 32, BB), 256, 0, stream>>>(
        xi, al + i * CC, be + i * CC, filt, xt);
    gemm512<<<dim3(TT / 32, 1, BB), 256, 0, stream>>>(
        (const __bf16*)Wb, (const __bf16*)xt, xi, cb + i * CC, out, dils[i]);
  }
}
